// ECLGCNN_4226247819406
// MI455X (gfx1250) — compile-verified, matrix operand order checked by reading, not run on hardware
//
#include <hip/hip_runtime.h>
#include <hip/hip_bf16.h>
#include <math.h>

// ---------------- problem constants (match reference) ----------------
#define KCH     3
#define TSTEPS  12
#define HID     256
#define BATCH   4096
#define NNODES  32
#define FEAT    5
#define NGRAPH  (BATCH * TSTEPS)        // 49152
#define NNODE   (NGRAPH * NNODES)       // 1572864
#define EPER    256
#define XW      (NNODES * FEAT)         // 160
#define GW      (4 * HID)               // 1024
#define BN_EPS  1e-5f

typedef __bf16 bf16_t;
typedef __attribute__((ext_vector_type(16))) __bf16 v16bf;
typedef __attribute__((ext_vector_type(8)))  __bf16 v8bf;
typedef __attribute__((ext_vector_type(8)))  float  v8f;

// ---------------- workspace layout (bytes, 1MB-aligned) ----------------
#define OFF_OUT    ((size_t)0)                  // pre-BN out f32 [N*5] = 31.5MB
#define OFF_PART   ((size_t)32 << 20)           // per-graph BN partials [49152*10] f32
#define OFF_STATS  ((size_t)34 << 20)           // BN affine A[5], B[5]
#define OFF_XSEQ   ((size_t)35 << 20)           // bf16 [T][B][160] = 15.7MB
#define OFF_HA     ((size_t)51 << 20)           // bf16 [B][256] = 2MB (ping)
#define OFF_HB     ((size_t)52 << 20)           // bf16 [B][256] = 2MB (pong)
#define OFF_C      ((size_t)53 << 20)           // f32  [B][256] = 4MB
#define OFF_YS     ((size_t)57 << 20)           // bf16 [B][T*256] = 25.2MB
#define OFF_WIH    ((size_t)81 << 20)           // bf16 [1024][160]
#define OFF_WHH    ((size_t)82 << 20)           // bf16 [1024][256]
#define OFF_BIAS   ((size_t)83 << 20)           // f32  [1024] (b_ih+b_hh)

__device__ __forceinline__ float sigf(float x) { return 1.f / (1.f + expf(-x)); }

// ============ 1) ChebConv per graph (one block = one graph) ============
__global__ void __launch_bounds__(256)
k_cheb(const float* __restrict__ x, const float* __restrict__ ea,
       const int* __restrict__ erow, const int* __restrict__ ecol,
       const float* __restrict__ wcheb,
       float* __restrict__ outp, float* __restrict__ part)
{
    __shared__ float xl[NNODES * FEAT], t1[NNODES * FEAT], t2[NNODES * FEAT];
    __shared__ float deg[NNODES], dis[NNODES];
    __shared__ float wS[KCH * FEAT * FEAT];
    __shared__ float acc[FEAT], accsq[FEAT];
    const int g   = blockIdx.x;
    const int tid = threadIdx.x;
    const int n0  = g * NNODES;

    if (tid < NNODES * FEAT) { xl[tid] = x[(size_t)n0 * FEAT + tid]; t1[tid] = 0.f; t2[tid] = 0.f; }
    if (tid < NNODES) deg[tid] = 0.f;
    if (tid < KCH * FEAT * FEAT) wS[tid] = wcheb[tid];
    if (tid < FEAT) { acc[tid] = 0.f; accsq[tid] = 0.f; }
    __syncthreads();

    const int ge = g * EPER + tid;                 // one thread per edge
    const int rl = erow[ge] - n0, cl = ecol[ge] - n0;
    const float at = ea[ge];
    atomicAdd(&deg[rl], at);
    __syncthreads();

    if (tid < NNODES) { float d = deg[tid]; dis[tid] = (d > 0.f) ? rsqrtf(d) : 0.f; }
    __syncthreads();

    const float nw = -dis[rl] * at * dis[cl];
    #pragma unroll
    for (int f = 0; f < FEAT; ++f) atomicAdd(&t1[cl * FEAT + f], nw * xl[rl * FEAT + f]);
    __syncthreads();
    #pragma unroll
    for (int f = 0; f < FEAT; ++f) atomicAdd(&t2[cl * FEAT + f], nw * t1[rl * FEAT + f]);
    __syncthreads();
    if (tid < NNODES * FEAT) t2[tid] = 2.f * t2[tid] - xl[tid];
    __syncthreads();

    if (tid < NNODES * FEAT) {
        const int n = tid / FEAT, fo = tid % FEAT;
        float o = 0.f;
        #pragma unroll
        for (int fi = 0; fi < FEAT; ++fi) {
            o += xl[n * FEAT + fi] * wS[ 0 + fi * FEAT + fo];
            o += t1[n * FEAT + fi] * wS[25 + fi * FEAT + fo];
            o += t2[n * FEAT + fi] * wS[50 + fi * FEAT + fo];
        }
        outp[(size_t)n0 * FEAT + tid] = o;
        atomicAdd(&acc[fo], o);
        atomicAdd(&accsq[fo], o * o);
    }
    __syncthreads();
    if (tid < FEAT) {
        part[(size_t)g * 10 + tid]     = acc[tid];
        part[(size_t)g * 10 + 5 + tid] = accsq[tid];
    }
}

// ============ 2) BN stats reduce -> affine A,B per feature ============
__global__ void k_bnstats(const float* __restrict__ part, const float* __restrict__ gamma,
                          const float* __restrict__ beta, float* __restrict__ stats)
{
    __shared__ float s[10];
    if (threadIdx.x < 10) s[threadIdx.x] = 0.f;
    __syncthreads();
    float loc[10];
    #pragma unroll
    for (int i = 0; i < 10; ++i) loc[i] = 0.f;
    for (int g = threadIdx.x; g < NGRAPH; g += blockDim.x)
        #pragma unroll
        for (int i = 0; i < 10; ++i) loc[i] += part[(size_t)g * 10 + i];
    #pragma unroll
    for (int i = 0; i < 10; ++i) atomicAdd(&s[i], loc[i]);
    __syncthreads();
    if (threadIdx.x < FEAT) {
        const int f = threadIdx.x;
        const float invn = 1.f / (float)NNODE;
        const float mean = s[f] * invn;
        const float var  = s[5 + f] * invn - mean * mean;   // biased variance
        const float A    = rsqrtf(var + BN_EPS) * gamma[f];
        stats[f]     = A;
        stats[5 + f] = beta[f] - mean * A;
    }
}

// ============ 3) apply BN + sigmoid, pack bf16 into [T][B][160] ============
__global__ void k_bnapply(const float* __restrict__ outp, const float* __restrict__ stats,
                          bf16_t* __restrict__ xseq)
{
    const size_t i = (size_t)blockIdx.x * blockDim.x + threadIdx.x;
    if (i >= (size_t)NNODE * FEAT) return;
    const int f = (int)(i % FEAT);
    const size_t n = i / FEAT;
    const float y = sigf(outp[i] * stats[f] + stats[5 + f]);
    const size_t graph = n / NNODES;
    const int nl = (int)(n % NNODES);
    const size_t b = graph / TSTEPS;
    const int t = (int)(graph % TSTEPS);
    xseq[((size_t)t * BATCH + b) * XW + nl * FEAT + f] = (bf16_t)y;
}

// ============ 4) weight conversion to bf16 + fused bias ============
__global__ void k_wprep(const float* __restrict__ wih, const float* __restrict__ whh,
                        const float* __restrict__ bih, const float* __restrict__ bhh,
                        bf16_t* __restrict__ wihb, bf16_t* __restrict__ whhb,
                        float* __restrict__ bias)
{
    const int i = blockIdx.x * blockDim.x + threadIdx.x;
    if (i < GW * XW)  wihb[i] = (bf16_t)wih[i];
    if (i < GW * HID) whhb[i] = (bf16_t)whh[i];
    if (i < GW)       bias[i] = bih[i] + bhh[i];
}

__global__ void k_zero(bf16_t* __restrict__ hb, float* __restrict__ cb)
{
    const int i = blockIdx.x * blockDim.x + threadIdx.x;
    if (i < BATCH * HID) { hb[i] = (bf16_t)0.f; cb[i] = 0.f; }
}

// ============ 5) fused LSTM step: WMMA GEMM (all 4 gates) + c/h update ============
// Wave: 16 batch-rows x 16 hidden-units, 4 accumulators (i,f,g,o) sharing one A
// fragment per K-chunk -> 13 chunks x 4 = 52 WMMAs/wave, bias folded into acc init.
// All four B fragments are loaded into distinct registers BEFORE the WMMAs so the
// loads issue as one clause and the waits stagger instead of serializing each WMMA.
// K = 416 = 5 chunks from Xt (stride 160) + 8 from H-in (stride 256).
// bf16 fragment layout (ISA 7.12.2): lane<16 -> K {0..7,16..23}; lane>=16 -> K {8..15,24..31}.
__global__ void __launch_bounds__(256)
k_lstm_step(const bf16_t* __restrict__ xseq_t, const bf16_t* __restrict__ hin,
            const bf16_t* __restrict__ wihb, const bf16_t* __restrict__ whhb,
            const float* __restrict__ bias,
            float* __restrict__ cbuf, bf16_t* __restrict__ hout,
            bf16_t* __restrict__ ys, int t)
{
    const int lane = threadIdx.x & 31;
    const int wv   = threadIdx.x >> 5;              // 8 waves: 2 (rows) x 4 (u-tiles)
    const int idx  = lane & 15;
    const int half = lane >> 4;
    const int row0 = blockIdx.x * 32 + (wv & 1) * 16;     // batch rows
    const int u0   = blockIdx.y * 64 + (wv >> 1) * 16;    // hidden units

    // bias folded into accumulator init (per gate, depends on column only)
    v8f accg[4];
    #pragma unroll
    for (int g = 0; g < 4; ++g) {
        const float bv = bias[g * HID + u0 + idx];
        #pragma unroll
        for (int r = 0; r < 8; ++r) accg[g][r] = bv;
    }

    #pragma unroll
    for (int c = 0; c < 13; ++c) {
        const bf16_t* aptr; const bf16_t* bptr; int stride, kk;
        if (c < 5) { aptr = xseq_t; bptr = wihb; stride = XW;  kk = c * 32; }
        else       { aptr = hin;    bptr = whhb; stride = HID; kk = (c - 5) * 32; }

        const bf16_t* ap = aptr + (size_t)(row0 + idx) * stride + kk + 8 * half;
        __builtin_prefetch(ap + 32, 0, 1);          // global_prefetch_b8
        const v8bf alo = *(const v8bf*)(ap);
        const v8bf ahi = *(const v8bf*)(ap + 16);
        v16bf A;
        #pragma unroll
        for (int j = 0; j < 8; ++j) { A[j] = alo[j]; A[8 + j] = ahi[j]; }

        // issue all 4 gate B-fragment loads first (distinct registers)
        v16bf Bm[4];
        #pragma unroll
        for (int g = 0; g < 4; ++g) {
            const bf16_t* bp = bptr + (size_t)(g * HID + u0 + idx) * stride + kk + 8 * half;
            const v8bf blo = *(const v8bf*)(bp);
            const v8bf bhi = *(const v8bf*)(bp + 16);
            #pragma unroll
            for (int j = 0; j < 8; ++j) { Bm[g][j] = blo[j]; Bm[g][8 + j] = bhi[j]; }
        }
        // then the 4 WMMAs, each reusing the same A fragment
        #pragma unroll
        for (int g = 0; g < 4; ++g)
            accg[g] = __builtin_amdgcn_wmma_f32_16x16x32_bf16(
                          false, A, false, Bm[g], (short)0, accg[g], false, false);
    }

    // gate nonlinearity + state update: lane owns (rows r+8*half, unit u0+idx)
    const int u = u0 + idx;
    #pragma unroll
    for (int r = 0; r < 8; ++r) {
        const int row = row0 + r + 8 * half;        // batch index
        const size_t cu = (size_t)row * HID + u;
        const float gi = sigf(accg[0][r]);
        const float gf = sigf(accg[1][r]);
        const float gg = tanhf(accg[2][r]);
        const float go = sigf(accg[3][r]);
        const float cc = gf * cbuf[cu] + gi * gg;
        cbuf[cu] = cc;
        const float h = go * tanhf(cc);
        hout[cu] = (bf16_t)h;
        ys[(size_t)row * (TSTEPS * HID) + (size_t)t * HID + u] = (bf16_t)sigf(h);
    }
}

// ============ 6) final linear [B,3072] x [4,3072]^T + sigmoid ============
__global__ void k_final(const bf16_t* __restrict__ ys, const float* __restrict__ wlin,
                        const float* __restrict__ blin, float* __restrict__ out)
{
    const int lane = threadIdx.x & 31;
    const int gw = blockIdx.x * (blockDim.x >> 5) + (threadIdx.x >> 5);
    const int b = gw >> 2, j = gw & 3;
    if (b >= BATCH) return;
    const bf16_t* yr = ys + (size_t)b * (TSTEPS * HID);
    const float* wr = wlin + (size_t)j * (TSTEPS * HID);
    float s = 0.f;
    for (int k = lane; k < TSTEPS * HID; k += 32) s += (float)yr[k] * wr[k];
    #pragma unroll
    for (int off = 16; off > 0; off >>= 1) s += __shfl_xor(s, off, 32);
    if (lane == 0) out[(size_t)b * 4 + j] = sigf(s + blin[j]);
}

// =====================================================================
extern "C" void kernel_launch(void* const* d_in, const int* in_sizes, int n_in,
                              void* d_out, int out_size, void* d_ws, size_t ws_size,
                              hipStream_t stream)
{
    (void)in_sizes; (void)n_in; (void)out_size; (void)ws_size;
    const float* x     = (const float*)d_in[0];
    const float* eattr = (const float*)d_in[1];
    const float* wcheb = (const float*)d_in[2];
    const float* gamma = (const float*)d_in[3];
    const float* beta  = (const float*)d_in[4];
    const float* wih   = (const float*)d_in[5];
    const float* whh   = (const float*)d_in[6];
    const float* bih   = (const float*)d_in[7];
    const float* bhh   = (const float*)d_in[8];
    const float* wlin  = (const float*)d_in[9];
    const float* blin  = (const float*)d_in[10];
    const int*   eidx  = (const int*)d_in[11];
    float* out = (float*)d_out;

    char* ws = (char*)d_ws;
    float*  outp  = (float*)(ws + OFF_OUT);
    float*  part  = (float*)(ws + OFF_PART);
    float*  stats = (float*)(ws + OFF_STATS);
    bf16_t* xseq  = (bf16_t*)(ws + OFF_XSEQ);
    bf16_t* hbufA = (bf16_t*)(ws + OFF_HA);
    bf16_t* hbufB = (bf16_t*)(ws + OFF_HB);
    float*  cbuf  = (float*)(ws + OFF_C);
    bf16_t* ysb   = (bf16_t*)(ws + OFF_YS);
    bf16_t* wihb  = (bf16_t*)(ws + OFF_WIH);
    bf16_t* whhb  = (bf16_t*)(ws + OFF_WHH);
    float*  bias  = (float*)(ws + OFF_BIAS);

    const int* erow = eidx;
    const int* ecol = eidx + (size_t)NGRAPH * EPER;

    k_cheb<<<NGRAPH, 256, 0, stream>>>(x, eattr, erow, ecol, wcheb, outp, part);
    k_bnstats<<<1, 256, 0, stream>>>(part, gamma, beta, stats);
    k_bnapply<<<((size_t)NNODE * FEAT + 255) / 256, 256, 0, stream>>>(outp, stats, xseq);
    k_wprep<<<(GW * HID + 255) / 256, 256, 0, stream>>>(wih, whh, bih, bhh, wihb, whhb, bias);
    k_zero<<<(BATCH * HID + 255) / 256, 256, 0, stream>>>(hbufA, cbuf);

    for (int t = 0; t < TSTEPS; ++t) {
        const bf16_t* hin = (t & 1) ? hbufB : hbufA;   // ping-pong H (RAW-safe)
        bf16_t*      hout = (t & 1) ? hbufA : hbufB;
        k_lstm_step<<<dim3(BATCH / 32, HID / 64), 256, 0, stream>>>(
            xseq + (size_t)t * BATCH * XW, hin, wihb, whhb, bias, cbuf, hout, ysb, t);
    }

    k_final<<<(BATCH * 4) / 8, 256, 0, stream>>>(ysb, wlin, blin, out);
}
